// RouteNetRecurrentGate_25331717112314
// MI455X (gfx1250) — matrix operand order repaired and checked by reading, not compiled, not run on hardware
//
#include <hip/hip_runtime.h>
#include <hip/hip_bf16.h>

typedef __bf16 v16bf __attribute__((ext_vector_type(16)));
typedef __bf16 v8bf  __attribute__((ext_vector_type(8)));
typedef __bf16 v4bf  __attribute__((ext_vector_type(4)));
typedef float  v8f   __attribute__((ext_vector_type(8)));

#define BATCH   16384
#define NIN     784
#define NBANK   6
#define DDIM    64
#define FEAT    384      // 6 banks * 64
#define NOUT    10
#define EPS_BN  1e-5f

#define LDA     40       // bf16 elems per A row  (80 B: 16B aligned, bank-spreading)
#define LDBT    40       // bf16 elems per Bt row

// ---------------- ws layout (float offsets) ----------------
#define WS_IN_SUM    0          // 784
#define WS_IN_SQ     784        // 784
#define WS_IN_SCALE  1568       // 784
#define WS_IN_SHIFT  2352       // 784
#define WS_LSTAT     3136       // 6 layers * (sum 384, sq 384, scale 384, shift 384)
#define WS_LSTRIDE   1536
#define WS_H0        (3136 + 6*1536)          // 12352
#define WS_H1        (WS_H0 + BATCH*FEAT)
#define WS_ZERO_N    WS_H0                    // zero all stats each call

// ================= WMMA fragment helpers =================
// A 16x32 bf16: lane half selects K = 16*half + 0..15 (matches B fragment order);
// per lane that is one contiguous 16-elem run -> two b128 LDS loads.
__device__ __forceinline__ v16bf load_A_frag(const __bf16* As, int mi, int lane) {
  const int half = lane >> 4, m = lane & 15;
  const __bf16* rowp = As + (size_t)(mi * 16 + m) * LDA;
  const v8bf lo = *(const v8bf*)(rowp + 16 * half);       // k = 16*half + 0..7
  const v8bf hi = *(const v8bf*)(rowp + 16 * half + 8);   // k = 16*half + 8..15
  return __builtin_shufflevector(lo, hi, 0,1,2,3,4,5,6,7,8,9,10,11,12,13,14,15);
}
// B 32x16 bf16: per lane k = 16*half + 0..15 for fixed column (lane&15).
// Bt is stored transposed in LDS: Bt[col][k] -> contiguous, two b128 loads.
__device__ __forceinline__ v16bf load_B_frag(const __bf16* Bt, int ni, int lane) {
  const int half = lane >> 4, n = lane & 15;
  const __bf16* colp = Bt + (size_t)(ni * 16 + n) * LDBT + 16 * half;
  const v8bf lo = *(const v8bf*)(colp);
  const v8bf hi = *(const v8bf*)(colp + 8);
  return __builtin_shufflevector(lo, hi, 0,1,2,3,4,5,6,7,8,9,10,11,12,13,14,15);
}

// ================= utility kernels =================
__global__ void zero_kernel(float* __restrict__ p, int n) {
  int i = blockIdx.x * blockDim.x + threadIdx.x;
  if (i < n) p[i] = 0.0f;
}

__global__ __launch_bounds__(256)
void input_stats_kernel(const float* __restrict__ x, float* __restrict__ sum, float* __restrict__ sq) {
  const int tid = threadIdx.x;
  const int r0  = blockIdx.x * 256;          // 64 blocks * 256 rows = 16384
  float s[4] = {0.f, 0.f, 0.f, 0.f}, q[4] = {0.f, 0.f, 0.f, 0.f};
  for (int r = 0; r < 256; ++r) {
    const float* xr = x + (size_t)(r0 + r) * NIN;
#pragma unroll
    for (int j = 0; j < 4; ++j) {
      const int f = tid + j * 256;
      if (f < NIN) { const float v = xr[f]; s[j] += v; q[j] += v * v; }
    }
  }
#pragma unroll
  for (int j = 0; j < 4; ++j) {
    const int f = tid + j * 256;
    if (f < NIN) { atomicAdd(&sum[f], s[j]); atomicAdd(&sq[f], q[j]); }
  }
}

__global__ void bn_finalize_kernel(const float* __restrict__ sum, const float* __restrict__ sq,
                                   const float* __restrict__ gamma, const float* __restrict__ beta,
                                   float* __restrict__ scale, float* __restrict__ shift,
                                   int n, float invCnt) {
  const int i = blockIdx.x * blockDim.x + threadIdx.x;
  if (i >= n) return;
  const float mu   = sum[i] * invCnt;
  const float var  = sq[i] * invCnt - mu * mu;
  const float rstd = rsqrtf(var + EPS_BN);
  const float sc   = gamma[i] * rstd;
  scale[i] = sc;
  shift[i] = beta[i] - mu * sc;
}

// ================= input GEMM: h0 = relu(BN(x) @ W_in^T + b_in) + layer-0 BN stats =================
__global__ __launch_bounds__(256)
void input_gemm_kernel(const float* __restrict__ x,
                       const float* __restrict__ scaleIn, const float* __restrict__ shiftIn,
                       const float* __restrict__ W_in,   // [6][64][784]
                       const float* __restrict__ b_in,   // [6][64]
                       float* __restrict__ h0,           // [B][384]
                       float* __restrict__ statSum, float* __restrict__ statSq) {
  __shared__ alignas(16) __bf16 As[128 * LDA];
  __shared__ alignas(16) __bf16 Bt[64 * LDBT];
  __shared__ float sSum[64], sSq[64];

  const int tid = threadIdx.x, wid = tid >> 5, lane = tid & 31;
  const int bi = blockIdx.x, kbank = blockIdx.y;
  const int b0 = bi * 128;
  if (tid < 64) { sSum[tid] = 0.f; sSq[tid] = 0.f; }

  v8f acc[4];
#pragma unroll
  for (int ni = 0; ni < 4; ++ni)
#pragma unroll
    for (int r = 0; r < 8; ++r) acc[ni][r] = 0.f;

  const float* Wb = W_in + (size_t)kbank * DDIM * NIN;

  // ---- main K loop: 24 chunks of 32, fully in range -> branchless vector fills ----
  for (int c = 0; c < 24; ++c) {
    const int k0 = c * 32;
    __syncthreads();
#pragma unroll
    for (int it = 0; it < 4; ++it) {              // A: 128x32 normalized x -> bf16
      const int linear = it * 256 + tid;
      const int row = linear >> 3;
      const int kg  = (linear & 7) * 4;
      const int nf  = k0 + kg;
      const float4 xv = *(const float4*)(x + (size_t)(b0 + row) * NIN + nf);
      const float4 sc = *(const float4*)(scaleIn + nf);
      const float4 sh = *(const float4*)(shiftIn + nf);
      v4bf o;
      o[0] = (__bf16)(xv.x * sc.x + sh.x);
      o[1] = (__bf16)(xv.y * sc.y + sh.y);
      o[2] = (__bf16)(xv.z * sc.z + sh.z);
      o[3] = (__bf16)(xv.w * sc.w + sh.w);
      *(v4bf*)&As[row * LDA + kg] = o;
    }
#pragma unroll
    for (int it = 0; it < 2; ++it) {              // B: W_in[d][n] -> Bt[d][kk]
      const int linear = it * 256 + tid;
      const int d  = linear >> 3;
      const int kg = (linear & 7) * 4;
      const float4 w4 = *(const float4*)(Wb + (size_t)d * NIN + k0 + kg);
      v4bf o;
      o[0] = (__bf16)w4.x; o[1] = (__bf16)w4.y; o[2] = (__bf16)w4.z; o[3] = (__bf16)w4.w;
      *(v4bf*)&Bt[d * LDBT + kg] = o;
    }
    __syncthreads();
    const v16bf af = load_A_frag(As, wid, lane);
#pragma unroll
    for (int ni = 0; ni < 4; ++ni) {
      const v16bf bfrag = load_B_frag(Bt, ni, lane);
      acc[ni] = __builtin_amdgcn_wmma_f32_16x16x32_bf16(false, af, false, bfrag,
                                                        (short)0, acc[ni], false, false);
    }
  }

  // ---- tail chunk: k0 = 768; 784-768 = 16 = float4-aligned boundary, so each
  //      4-group is either fully valid (kg < 16) or fully zero -> still branch-lite ----
  {
    const int k0 = 24 * 32;
    __syncthreads();
#pragma unroll
    for (int it = 0; it < 4; ++it) {
      const int linear = it * 256 + tid;
      const int row = linear >> 3;
      const int kg  = (linear & 7) * 4;
      v4bf o;
      if (kg < 16) {
        const int nf = k0 + kg;
        const float4 xv = *(const float4*)(x + (size_t)(b0 + row) * NIN + nf);
        const float4 sc = *(const float4*)(scaleIn + nf);
        const float4 sh = *(const float4*)(shiftIn + nf);
        o[0] = (__bf16)(xv.x * sc.x + sh.x);
        o[1] = (__bf16)(xv.y * sc.y + sh.y);
        o[2] = (__bf16)(xv.z * sc.z + sh.z);
        o[3] = (__bf16)(xv.w * sc.w + sh.w);
      } else {
        o[0] = (__bf16)0.f; o[1] = (__bf16)0.f; o[2] = (__bf16)0.f; o[3] = (__bf16)0.f;
      }
      *(v4bf*)&As[row * LDA + kg] = o;
    }
#pragma unroll
    for (int it = 0; it < 2; ++it) {
      const int linear = it * 256 + tid;
      const int d  = linear >> 3;
      const int kg = (linear & 7) * 4;
      v4bf o;
      if (kg < 16) {
        const float4 w4 = *(const float4*)(Wb + (size_t)d * NIN + k0 + kg);
        o[0] = (__bf16)w4.x; o[1] = (__bf16)w4.y; o[2] = (__bf16)w4.z; o[3] = (__bf16)w4.w;
      } else {
        o[0] = (__bf16)0.f; o[1] = (__bf16)0.f; o[2] = (__bf16)0.f; o[3] = (__bf16)0.f;
      }
      *(v4bf*)&Bt[d * LDBT + kg] = o;
    }
    __syncthreads();
    const v16bf af = load_A_frag(As, wid, lane);
#pragma unroll
    for (int ni = 0; ni < 4; ++ni) {
      const v16bf bfrag = load_B_frag(Bt, ni, lane);
      acc[ni] = __builtin_amdgcn_wmma_f32_16x16x32_bf16(false, af, false, bfrag,
                                                        (short)0, acc[ni], false, false);
    }
  }

  const int half = lane >> 4, n = lane & 15;
#pragma unroll
  for (int ni = 0; ni < 4; ++ni) {
    const int e = ni * 16 + n;
    const float bias = b_in[kbank * DDIM + e];
#pragma unroll
    for (int r = 0; r < 8; ++r) {
      const int row = wid * 16 + r + 8 * half;
      float v = acc[ni][r] + bias;
      v = fmaxf(v, 0.f);
      h0[(size_t)(b0 + row) * FEAT + kbank * DDIM + e] = v;
      atomicAdd(&sSum[e], v);
      atomicAdd(&sSq[e], v * v);
    }
  }
  __syncthreads();
  if (tid < 64) {
    atomicAdd(&statSum[kbank * DDIM + tid], sSum[tid]);
    atomicAdd(&statSq[kbank * DDIM + tid], sSq[tid]);
  }
}

// ================= fused routing layer: gates + gated K=384 WMMA GEMM per target =================
__global__ __launch_bounds__(256)
void edge_layer_kernel(const float* __restrict__ h_prev,  // [B][384] pre-BN relu acts
                       const float* __restrict__ scale,   // [384] BN fold for this layer
                       const float* __restrict__ shift,
                       const float* __restrict__ Wg,      // [6][6][64]  (s,t,d)
                       const float* __restrict__ Wd,      // [6][6][64][64] (s,t,d,e)
                       const float* __restrict__ bd,      // [6][6][64]
                       float* __restrict__ h_next,
                       float* __restrict__ statSum, float* __restrict__ statSq,  // [384]
                       float* __restrict__ total_gate) {  // [B]
  __shared__ float  gateS[128][6];
  __shared__ alignas(16) __bf16 As[128 * LDA];
  __shared__ alignas(16) __bf16 Bt[64 * LDBT];
  __shared__ float  sSum[64], sSq[64];

  const int tid = threadIdx.x, wid = tid >> 5, lane = tid & 31;
  const int bi = blockIdx.x, t = blockIdx.y;
  const int b0 = bi * 128;
  if (tid < 64) { sSum[tid] = 0.f; sSq[tid] = 0.f; }

  // ---- gates: hardtanh(dot(actsN[b,s,:], Wg[s,t,:]), 0, 1), float4 dots ----
  for (int g = tid; g < 128 * 6; g += 256) {
    const int row = g / 6, s = g % 6;
    const float* hr = h_prev + (size_t)(b0 + row) * FEAT + s * DDIM;
    const float* sc = scale + s * DDIM;
    const float* sh = shift + s * DDIM;
    const float* wg = Wg + (size_t)(s * 6 + t) * DDIM;
    float a = 0.f;
#pragma unroll
    for (int d4 = 0; d4 < DDIM / 4; ++d4) {
      const float4 h4 = *(const float4*)(hr + d4 * 4);
      const float4 s4 = *(const float4*)(sc + d4 * 4);
      const float4 b4 = *(const float4*)(sh + d4 * 4);
      const float4 w4 = *(const float4*)(wg + d4 * 4);
      a += (h4.x * s4.x + b4.x) * w4.x + (h4.y * s4.y + b4.y) * w4.y +
           (h4.z * s4.z + b4.z) * w4.z + (h4.w * s4.w + b4.w) * w4.w;
    }
    gateS[row][s] = fminf(fmaxf(a, 0.f), 1.f);
  }
  __syncthreads();
  if (tid < 128) {
    const float gsum = gateS[tid][0] + gateS[tid][1] + gateS[tid][2] +
                       gateS[tid][3] + gateS[tid][4] + gateS[tid][5];
    atomicAdd(&total_gate[b0 + tid], gsum);
  }

  v8f acc[4];
#pragma unroll
  for (int ni = 0; ni < 4; ++ni)
#pragma unroll
    for (int r = 0; r < 8; ++r) acc[ni][r] = 0.f;

  // ---- gated GEMM: pre[b,e] = sum_{s,d} (gate[b,s]*actsN[b,s,d]) * Wd[s,t,d,e] ----
  for (int s = 0; s < 6; ++s) {
    for (int c = 0; c < 2; ++c) {
      __syncthreads();
      // A: 128x32 gated+normalized -> bf16, float4 loads + v4bf stores
#pragma unroll
      for (int it = 0; it < 4; ++it) {
        const int linear = it * 256 + tid;
        const int row = linear >> 3;
        const int kg  = (linear & 7) * 4;
        const int f   = s * DDIM + c * 32 + kg;
        const float4 h4 = *(const float4*)(h_prev + (size_t)(b0 + row) * FEAT + f);
        const float4 s4 = *(const float4*)(scale + f);
        const float4 b4 = *(const float4*)(shift + f);
        const float  g  = gateS[row][s];
        v4bf o;
        o[0] = (__bf16)((h4.x * s4.x + b4.x) * g);
        o[1] = (__bf16)((h4.y * s4.y + b4.y) * g);
        o[2] = (__bf16)((h4.z * s4.z + b4.z) * g);
        o[3] = (__bf16)((h4.w * s4.w + b4.w) * g);
        *(v4bf*)&As[row * LDA + kg] = o;
      }
      // B: Wd[d][e] chunk -> transposed Bt[e][kk] (coalesced float4 over e)
      const float* wd = Wd + ((size_t)(s * 6 + t) * DDIM + c * 32) * DDIM;
#pragma unroll
      for (int it = 0; it < 2; ++it) {
        const int linear = it * 256 + tid;          // 512 groups (32 kk x 16 egroups)
        const int kk = linear >> 4;
        const int eg = (linear & 15) * 4;
        const float4 w4 = *(const float4*)(wd + (size_t)kk * DDIM + eg);
        Bt[(eg + 0) * LDBT + kk] = (__bf16)w4.x;
        Bt[(eg + 1) * LDBT + kk] = (__bf16)w4.y;
        Bt[(eg + 2) * LDBT + kk] = (__bf16)w4.z;
        Bt[(eg + 3) * LDBT + kk] = (__bf16)w4.w;
      }
      __syncthreads();
      const v16bf af = load_A_frag(As, wid, lane);
#pragma unroll
      for (int ni = 0; ni < 4; ++ni) {
        const v16bf bfrag = load_B_frag(Bt, ni, lane);
        acc[ni] = __builtin_amdgcn_wmma_f32_16x16x32_bf16(false, af, false, bfrag,
                                                          (short)0, acc[ni], false, false);
      }
    }
  }

  // ---- epilogue: += sum_s gate*bd, relu, store, BN stats ----
  const int half = lane >> 4, n = lane & 15;
#pragma unroll
  for (int ni = 0; ni < 4; ++ni) {
    const int e = ni * 16 + n;
    float bdv[6];
#pragma unroll
    for (int s = 0; s < 6; ++s) bdv[s] = bd[(size_t)(s * 6 + t) * DDIM + e];
#pragma unroll
    for (int r = 0; r < 8; ++r) {
      const int row = wid * 16 + r + 8 * half;
      float bias = 0.f;
#pragma unroll
      for (int s = 0; s < 6; ++s) bias += gateS[row][s] * bdv[s];
      float v = acc[ni][r] + bias;
      v = fmaxf(v, 0.f);
      h_next[(size_t)(b0 + row) * FEAT + t * DDIM + e] = v;
      atomicAdd(&sSum[e], v);
      atomicAdd(&sSq[e], v * v);
    }
  }
  __syncthreads();
  if (tid < 64) {
    atomicAdd(&statSum[t * DDIM + tid], sSum[tid]);
    atomicAdd(&statSq[t * DDIM + tid], sSq[tid]);
  }
}

// ================= output projection =================
__global__ __launch_bounds__(256)
void output_kernel(const float* __restrict__ h5, const float* __restrict__ scale,
                   const float* __restrict__ shift, const float* __restrict__ W_out,
                   float* __restrict__ out) {
  const int b = blockIdx.x * 256 + threadIdx.x;
  float acc[NOUT];
#pragma unroll
  for (int o = 0; o < NOUT; ++o) acc[o] = 0.f;
  for (int f = 0; f < FEAT; f += 4) {
    const float4 h4 = *(const float4*)(h5 + (size_t)b * FEAT + f);
    const float4 s4 = *(const float4*)(scale + f);
    const float4 b4 = *(const float4*)(shift + f);
    const float a0 = h4.x * s4.x + b4.x, a1 = h4.y * s4.y + b4.y;
    const float a2 = h4.z * s4.z + b4.z, a3 = h4.w * s4.w + b4.w;
    const int k = f >> 6, d = f & 63;
    const float* w = W_out + (size_t)k * NOUT * DDIM + d;
#pragma unroll
    for (int o = 0; o < NOUT; ++o) {
      const float4 w4 = *(const float4*)(w + o * DDIM);
      acc[o] += a0 * w4.x + a1 * w4.y + a2 * w4.z + a3 * w4.w;
    }
  }
#pragma unroll
  for (int o = 0; o < NOUT; ++o) out[(size_t)b * NOUT + o] = acc[o];
}

// ================= launch =================
extern "C" void kernel_launch(void* const* d_in, const int* in_sizes, int n_in,
                              void* d_out, int out_size, void* d_ws, size_t ws_size,
                              hipStream_t stream) {
  (void)in_sizes; (void)n_in; (void)out_size; (void)ws_size;
  const float* x        = (const float*)d_in[0];
  const float* gamma_in = (const float*)d_in[1];
  const float* beta_in  = (const float*)d_in[2];
  const float* W_in     = (const float*)d_in[3];
  const float* b_in     = (const float*)d_in[4];
  const float* Wg       = (const float*)d_in[5];
  const float* Wd       = (const float*)d_in[6];
  const float* bd       = (const float*)d_in[7];
  const float* gamma_h  = (const float*)d_in[8];
  const float* beta_h   = (const float*)d_in[9];
  const float* W_out    = (const float*)d_in[10];

  float* ws  = (float*)d_ws;
  float* out = (float*)d_out;                      // [B*10]
  float* total_gate = out + (size_t)BATCH * NOUT;  // [B]

  float* hA = ws + WS_H0;
  float* hB = ws + WS_H1;
  const float invB = 1.0f / (float)BATCH;

  zero_kernel<<<(WS_ZERO_N + 255) / 256, 256, 0, stream>>>(ws, WS_ZERO_N);
  zero_kernel<<<(BATCH + 255) / 256, 256, 0, stream>>>(total_gate, BATCH);

  input_stats_kernel<<<BATCH / 256, 256, 0, stream>>>(x, ws + WS_IN_SUM, ws + WS_IN_SQ);
  bn_finalize_kernel<<<(NIN + 255) / 256, 256, 0, stream>>>(
      ws + WS_IN_SUM, ws + WS_IN_SQ, gamma_in, beta_in,
      ws + WS_IN_SCALE, ws + WS_IN_SHIFT, NIN, invB);

  {
    float* st = ws + WS_LSTAT + 0 * WS_LSTRIDE;
    dim3 grid(BATCH / 128, NBANK);
    input_gemm_kernel<<<grid, 256, 0, stream>>>(
        x, ws + WS_IN_SCALE, ws + WS_IN_SHIFT, W_in, b_in, hA, st, st + FEAT);
  }

  float* hsrc = hA;
  float* hdst = hB;
  for (int l = 0; l < 5; ++l) {
    float* stl = ws + WS_LSTAT + (size_t)l * WS_LSTRIDE;
    float* stn = ws + WS_LSTAT + (size_t)(l + 1) * WS_LSTRIDE;
    bn_finalize_kernel<<<(FEAT + 255) / 256, 256, 0, stream>>>(
        stl, stl + FEAT, gamma_h + (size_t)l * FEAT, beta_h + (size_t)l * FEAT,
        stl + 2 * FEAT, stl + 3 * FEAT, FEAT, invB);
    dim3 grid(BATCH / 128, NBANK);
    edge_layer_kernel<<<grid, 256, 0, stream>>>(
        hsrc, stl + 2 * FEAT, stl + 3 * FEAT,
        Wg + (size_t)l * NBANK * NBANK * DDIM,
        Wd + (size_t)l * NBANK * NBANK * DDIM * DDIM,
        bd + (size_t)l * NBANK * NBANK * DDIM,
        hdst, stn, stn + FEAT, total_gate);
    float* tmp = hsrc; hsrc = hdst; hdst = tmp;
  }

  {
    float* st5 = ws + WS_LSTAT + 5 * WS_LSTRIDE;
    bn_finalize_kernel<<<(FEAT + 255) / 256, 256, 0, stream>>>(
        st5, st5 + FEAT, gamma_h + 5 * FEAT, beta_h + 5 * FEAT,
        st5 + 2 * FEAT, st5 + 3 * FEAT, FEAT, invB);
    output_kernel<<<BATCH / 256, 256, 0, stream>>>(
        hsrc, st5 + 2 * FEAT, st5 + 3 * FEAT, W_out, out);
  }
}
